// GraphAttentionBlock_61340722922151
// MI455X (gfx1250) — matrix operand order repaired and checked by reading, not compiled, not run on hardware
//
#include <hip/hip_runtime.h>
#include <math.h>

// ---------------------------------------------------------------------------
// GraphAttentionBlock for MI455X (gfx1250, wave32, WMMA).
// All GEMM-shaped work runs on V_WMMA_F32_16X16X32_BF16 (bf16 in, f32 acc).
// Attention is flash-style streaming with the adjacency bias + post-softmax
// re-mask folded into the online-softmax recurrence. V is pre-transposed to
// [b][h][d][n] so every WMMA fragment load is contiguous-dword (b128-able).
// ---------------------------------------------------------------------------

typedef __attribute__((ext_vector_type(8)))  float        v8f;
typedef __attribute__((ext_vector_type(16))) __bf16       v16bf;
typedef __attribute__((ext_vector_type(8)))  unsigned int v8u;
typedef unsigned short u16;
typedef unsigned int   u32;

#define D_MODEL 512
#define SEQ_N   2048
#define BATCH   4
#define NHEADS  8
#define HEADD   64
#define FFDIM   2048
#define MROWS   (BATCH * SEQ_N)   // 8192

__device__ __forceinline__ u16 f2bf(float f) {
    u32 u = __float_as_uint(f);
    u = u + 0x7FFFu + ((u >> 16) & 1u);   // round-to-nearest-even
    return (u16)(u >> 16);
}

// ---------------------------------------------------------------------------
// Prep kernels
// ---------------------------------------------------------------------------
__global__ void adj_to_u8_kernel(const float* __restrict__ adj,
                                 unsigned char* __restrict__ out, int n) {
    int i = blockIdx.x * blockDim.x + threadIdx.x;
    if (i < n) out[i] = adj[i] > 0.5f ? 1 : 0;
}

// W[K][N] (row-major, as in reference) -> Wt[N][K] bf16, so WMMA B-fragments
// read contiguous (k,k+1) pairs as single dwords.
__global__ void transpose_to_bf16_kernel(const float* __restrict__ W,
                                         u16* __restrict__ Wt, int K, int N) {
    int i = blockIdx.x * blockDim.x + threadIdx.x;
    if (i >= K * N) return;
    int k = i / N, n = i % N;
    Wt[(size_t)n * K + k] = f2bf(W[i]);
}

// v_bf [B*N][512] -> vT [B][H][64][SEQ_N], so attention V-fragments read
// contiguous dword pairs along the key (contraction) axis.
__global__ void v_transpose_kernel(const u16* __restrict__ v,
                                   u16* __restrict__ vT) {
    int i = blockIdx.x * blockDim.x + threadIdx.x;
    if (i >= MROWS * D_MODEL) return;
    int row = i >> 9;          // b*SEQ_N + n
    int col = i & 511;         // h*64 + d
    int b = row / SEQ_N, n = row % SEQ_N;
    int h = col >> 6, d = col & 63;
    vT[(size_t)((b * NHEADS + h) * HEADD + d) * SEQ_N + n] = v[i];
}

// Per-row LayerNorm over D=512, output bf16. One block (256 threads) per row.
__global__ __launch_bounds__(256)
void layernorm_bf16_kernel(const float* __restrict__ x,
                           const float* __restrict__ g,
                           const float* __restrict__ b,
                           u16* __restrict__ out) {
    const int D = D_MODEL;
    size_t base = (size_t)blockIdx.x * D;
    int t = threadIdx.x;
    float x0 = x[base + t], x1 = x[base + t + 256];
    __shared__ float ssum[256], ssq[256];
    ssum[t] = x0 + x1;
    ssq[t]  = x0 * x0 + x1 * x1;
    __syncthreads();
    for (int s = 128; s > 0; s >>= 1) {
        if (t < s) { ssum[t] += ssum[t + s]; ssq[t] += ssq[t + s]; }
        __syncthreads();
    }
    float mean = ssum[0] * (1.0f / D);
    float var  = ssq[0] * (1.0f / D) - mean * mean;
    float rstd = rsqrtf(var + 1e-5f);
    out[base + t]       = f2bf((x0 - mean) * rstd * g[t]       + b[t]);
    out[base + t + 256] = f2bf((x1 - mean) * rstd * g[t + 256] + b[t + 256]);
}

// ---------------------------------------------------------------------------
// Generic WMMA GEMM: C[M][Nn] = A_bf16[M][K] @ B (given transposed Bt[Nn][K])
//                     + bias, optional exact GELU, optional residual add.
// Each wave owns a 32x64 output tile: 2x4 grid of 16x16 WMMA accumulators.
// ---------------------------------------------------------------------------
template <bool GELU, bool RESID, bool OUT_BF16>
__global__ __launch_bounds__(256)
void gemm_bf16_kernel(const u16* __restrict__ A, const u16* __restrict__ Bt,
                      const float* __restrict__ bias,
                      const float* __restrict__ resid, void* __restrict__ out,
                      int M, int K, int Nn) {
    const int wavesPerBlk = blockDim.x >> 5;
    int wave = blockIdx.x * wavesPerBlk + (threadIdx.x >> 5);
    int lane = threadIdx.x & 31;
    int mtiles = M >> 5, ntiles = Nn >> 6;
    if (wave >= mtiles * ntiles) return;
    int m0 = (wave / ntiles) << 5;
    int n0 = (wave % ntiles) << 6;
    int rl = lane & 15;        // row (A) / col (B/C) within tile
    int hv = lane >> 4;        // lane half selects K-subblock / M-subblock

    v8f acc[2][4] = {};

    for (int k0 = 0; k0 < K; k0 += 32) {
        // A fragments: 16x32 bf16, lanes 0-15 hold K=0..15, lanes 16-31 K=16..31.
        v16bf af[2];
#pragma unroll
        for (int i = 0; i < 2; ++i) {
            const u16* ap = A + (size_t)(m0 + i * 16 + rl) * K + k0 + hv * 16;
            v8u u;
            u[0] = *(const u32*)(ap + 0);  u[1] = *(const u32*)(ap + 2);
            u[2] = *(const u32*)(ap + 4);  u[3] = *(const u32*)(ap + 6);
            u[4] = *(const u32*)(ap + 8);  u[5] = *(const u32*)(ap + 10);
            u[6] = *(const u32*)(ap + 12); u[7] = *(const u32*)(ap + 14);
            af[i] = __builtin_bit_cast(v16bf, u);
        }
        // B fragments: 32x16 bf16, lanes 0-15 hold K=0..15, lanes 16-31 K=16..31.
        v16bf bf[4];
#pragma unroll
        for (int i = 0; i < 4; ++i) {
            const u16* bp = Bt + (size_t)(n0 + i * 16 + rl) * K + k0 + hv * 16;
            v8u u;
#pragma unroll
            for (int j = 0; j < 8; ++j) u[j] = *(const u32*)(bp + 2 * j);
            bf[i] = __builtin_bit_cast(v16bf, u);
        }
#pragma unroll
        for (int i = 0; i < 2; ++i)
#pragma unroll
            for (int j = 0; j < 4; ++j)
                acc[i][j] = __builtin_amdgcn_wmma_f32_16x16x32_bf16(
                    false, af[i], false, bf[j], (short)0, acc[i][j], false, false);
    }

    // C layout: VGPR r holds row M = hv*8 + r, column = lane&15.
#pragma unroll
    for (int i = 0; i < 2; ++i)
#pragma unroll
        for (int j = 0; j < 4; ++j) {
            int c = n0 + j * 16 + rl;
            float bv = bias[c];
#pragma unroll
            for (int r = 0; r < 8; ++r) {
                int row = m0 + i * 16 + hv * 8 + r;
                float x = acc[i][j][r] + bv;
                if (GELU) x = 0.5f * x * (1.0f + erff(x * 0.70710678118f));
                size_t idx = (size_t)row * Nn + c;
                if (RESID) x += resid[idx];
                if (OUT_BF16) ((u16*)out)[idx] = f2bf(x);
                else          ((float*)out)[idx] = x;
            }
        }
}

// ---------------------------------------------------------------------------
// Flash-style graph attention. One wave = 16 queries of one (batch, head).
// logits = (QK^T)/8 + (2a-1); post-softmax mask m=0.25+0.75a and renorm
// reduce to: O = sum(e_i * m_i * V_i) / sum(e_i * m_i)  (clamp never fires,
// since sum(e*m) >= 0.25 * sum(e) >> 1e-6 * sum(e)).
// ---------------------------------------------------------------------------
__global__ __launch_bounds__(256)
void graph_attention_kernel(const u16* __restrict__ qbf,
                            const u16* __restrict__ kbf,
                            const u16* __restrict__ vT,
                            const unsigned char* __restrict__ adj,
                            u16* __restrict__ attbf) {
    __shared__ __attribute__((aligned(16))) u16 plds[8][16][32];
    const int wavesPerBlk = blockDim.x >> 5;
    int wave = blockIdx.x * wavesPerBlk + (threadIdx.x >> 5);
    int wl   = threadIdx.x >> 5;
    int lane = threadIdx.x & 31;
    const int qtilesPerBH = SEQ_N / 16;                 // 128
    int bh = wave / qtilesPerBH;                        // 0..31
    int qt = wave % qtilesPerBH;
    int b = bh / NHEADS, h = bh % NHEADS;
    int q0 = qt * 16;
    int rl = lane & 15;
    int hv = lane >> 4;

    // Q fragments (16 x 64 split into two 16x32 k-steps), A layout.
    v16bf qa[2];
#pragma unroll
    for (int i = 0; i < 2; ++i) {
        const u16* qp = qbf + (size_t)(b * SEQ_N + q0 + rl) * D_MODEL
                            + h * HEADD + 32 * i + hv * 16;
        v8u u;
        u[0] = *(const u32*)(qp + 0);  u[1] = *(const u32*)(qp + 2);
        u[2] = *(const u32*)(qp + 4);  u[3] = *(const u32*)(qp + 6);
        u[4] = *(const u32*)(qp + 8);  u[5] = *(const u32*)(qp + 10);
        u[6] = *(const u32*)(qp + 12); u[7] = *(const u32*)(qp + 14);
        qa[i] = __builtin_bit_cast(v16bf, u);
    }

    v8f o[4] = {};
    float mrow[8], lrow[8];
#pragma unroll
    for (int r = 0; r < 8; ++r) { mrow[r] = -1e30f; lrow[r] = 0.0f; }

    for (int kc = 0; kc < SEQ_N; kc += 32) {
        // ---- S = Q K^T for 32 keys (two 16x16 tiles) ----
        v8f s[2] = {};
#pragma unroll
        for (int c = 0; c < 2; ++c) {
#pragma unroll
            for (int i = 0; i < 2; ++i) {
                // B = K^T: B[kk][key] = Kmat[key][kk]; pairs contiguous along kk.
                const u16* kp = kbf + (size_t)(b * SEQ_N + kc + c * 16 + rl) * D_MODEL
                                    + h * HEADD + 32 * i + hv * 16;
                v8u u;
#pragma unroll
                for (int j = 0; j < 8; ++j) u[j] = *(const u32*)(kp + 2 * j);
                v16bf kf = __builtin_bit_cast(v16bf, u);
                s[c] = __builtin_amdgcn_wmma_f32_16x16x32_bf16(
                    false, qa[i], false, kf, (short)0, s[c], false, false);
            }
        }

        // ---- online softmax update (per query row = hv*8 + r) ----
        const unsigned char* abase = adj + (size_t)b * SEQ_N * SEQ_N + kc;
#pragma unroll
        for (int r = 0; r < 8; ++r) {
            int query = q0 + hv * 8 + r;
            const unsigned char* arow = abase + (size_t)query * SEQ_N;
            float a0 = (float)arow[rl];
            float a1 = (float)arow[16 + rl];
            float l0 = s[0][r] * 0.125f + 2.0f * a0 - 1.0f;
            float l1 = s[1][r] * 0.125f + 2.0f * a1 - 1.0f;
            float cm = fmaxf(l0, l1);
            cm = fmaxf(cm, __shfl_xor(cm, 1));
            cm = fmaxf(cm, __shfl_xor(cm, 2));
            cm = fmaxf(cm, __shfl_xor(cm, 4));
            cm = fmaxf(cm, __shfl_xor(cm, 8));     // stays within 16-lane half
            float mnew = fmaxf(mrow[r], cm);
            float corr = __expf(mrow[r] - mnew);
            float p0 = __expf(l0 - mnew) * (0.25f + 0.75f * a0);
            float p1 = __expf(l1 - mnew) * (0.25f + 0.75f * a1);
            float rs = p0 + p1;
            rs += __shfl_xor(rs, 1); rs += __shfl_xor(rs, 2);
            rs += __shfl_xor(rs, 4); rs += __shfl_xor(rs, 8);
            lrow[r] = lrow[r] * corr + rs;
            mrow[r] = mnew;
            o[0][r] *= corr; o[1][r] *= corr; o[2][r] *= corr; o[3][r] *= corr;
            int rowm = hv * 8 + r;
            plds[wl][rowm][rl]      = f2bf(p0);
            plds[wl][rowm][16 + rl] = f2bf(p1);
        }
        asm volatile("s_wait_dscnt 0x0" ::: "memory");   // P staged in LDS

        // ---- reshape P (C layout) -> A layout via LDS ----
        {
            const u16* pl = &plds[wl][rl][0];
            v8u u;
            u[0] = *(const u32*)(pl + hv * 16 + 0);
            u[1] = *(const u32*)(pl + hv * 16 + 2);
            u[2] = *(const u32*)(pl + hv * 16 + 4);
            u[3] = *(const u32*)(pl + hv * 16 + 6);
            u[4] = *(const u32*)(pl + hv * 16 + 8);
            u[5] = *(const u32*)(pl + hv * 16 + 10);
            u[6] = *(const u32*)(pl + hv * 16 + 12);
            u[7] = *(const u32*)(pl + hv * 16 + 14);
            v16bf pa = __builtin_bit_cast(v16bf, u);

            // ---- O += P @ V (V: 32 keys x 64 cols, 4 column tiles) ----
            // vT[b][h][d][n]: pairs along keys are contiguous -> dword loads.
#pragma unroll
            for (int j = 0; j < 4; ++j) {
                const u16* vp = vT + (size_t)((b * NHEADS + h) * HEADD + j * 16 + rl) * SEQ_N
                                   + kc + hv * 16;
                v8u u2;
#pragma unroll
                for (int jj = 0; jj < 8; ++jj)
                    u2[jj] = *(const u32*)(vp + 2 * jj);
                v16bf vf = __builtin_bit_cast(v16bf, u2);
                o[j] = __builtin_amdgcn_wmma_f32_16x16x32_bf16(
                    false, pa, false, vf, (short)0, o[j], false, false);
            }
        }
    }

    // ---- normalize and store attended (bf16, feeds the Wo GEMM) ----
#pragma unroll
    for (int r = 0; r < 8; ++r) {
        float inv = 1.0f / lrow[r];                 // sum(e*m) >= 0.25 always
        int query = q0 + hv * 8 + r;
        u16* op = attbf + (size_t)(b * SEQ_N + query) * D_MODEL + h * HEADD + rl;
        op[0]  = f2bf(o[0][r] * inv);
        op[16] = f2bf(o[1][r] * inv);
        op[32] = f2bf(o[2][r] * inv);
        op[48] = f2bf(o[3][r] * inv);
    }
}

// ---------------------------------------------------------------------------
// Host-side orchestration
// ---------------------------------------------------------------------------
extern "C" void kernel_launch(void* const* d_in, const int* in_sizes, int n_in,
                              void* d_out, int out_size, void* d_ws, size_t ws_size,
                              hipStream_t stream) {
    const float* hidden = (const float*)d_in[0];
    const float* adjf   = (const float*)d_in[1];
    const float* Wq  = (const float*)d_in[2];  const float* bq  = (const float*)d_in[3];
    const float* Wk  = (const float*)d_in[4];  const float* bk  = (const float*)d_in[5];
    const float* Wv  = (const float*)d_in[6];  const float* bv  = (const float*)d_in[7];
    const float* Wo  = (const float*)d_in[8];  const float* bo  = (const float*)d_in[9];
    const float* g1  = (const float*)d_in[10]; const float* b1  = (const float*)d_in[11];
    const float* g2  = (const float*)d_in[12]; const float* b2  = (const float*)d_in[13];
    const float* Wf1 = (const float*)d_in[14]; const float* bf1 = (const float*)d_in[15];
    const float* Wf2 = (const float*)d_in[16]; const float* bf2 = (const float*)d_in[17];

    char* ws = (char*)d_ws;
    size_t off = 0;
    auto alloc = [&](size_t bytes) -> void* {
        void* p = ws + off;
        off = (off + bytes + 255) & ~(size_t)255;
        return p;
    };
    u16* normed_bf  = (u16*)alloc((size_t)MROWS * D_MODEL * 2);
    u16* q_bf       = (u16*)alloc((size_t)MROWS * D_MODEL * 2);
    u16* k_bf       = (u16*)alloc((size_t)MROWS * D_MODEL * 2);
    u16* v_bf       = (u16*)alloc((size_t)MROWS * D_MODEL * 2);
    u16* vT_bf      = (u16*)alloc((size_t)MROWS * D_MODEL * 2);
    u16* att_bf     = (u16*)alloc((size_t)MROWS * D_MODEL * 2);
    float* hidden2  = (float*)alloc((size_t)MROWS * D_MODEL * 4);
    u16* normed2_bf = (u16*)alloc((size_t)MROWS * D_MODEL * 2);
    u16* ff1_bf     = (u16*)alloc((size_t)MROWS * FFDIM * 2);
    unsigned char* adj8 = (unsigned char*)alloc((size_t)BATCH * SEQ_N * SEQ_N);
    u16* wqT  = (u16*)alloc((size_t)D_MODEL * D_MODEL * 2);
    u16* wkT  = (u16*)alloc((size_t)D_MODEL * D_MODEL * 2);
    u16* wvT  = (u16*)alloc((size_t)D_MODEL * D_MODEL * 2);
    u16* woT  = (u16*)alloc((size_t)D_MODEL * D_MODEL * 2);
    u16* wf1T = (u16*)alloc((size_t)D_MODEL * FFDIM * 2);   // [FFDIM][D_MODEL]
    u16* wf2T = (u16*)alloc((size_t)FFDIM * D_MODEL * 2);   // [D_MODEL][FFDIM]

    // --- prep ---
    {
        int n = BATCH * SEQ_N * SEQ_N;
        adj_to_u8_kernel<<<(n + 255) / 256, 256, 0, stream>>>(adjf, adj8, n);
    }
    {
        int n = D_MODEL * D_MODEL;
        transpose_to_bf16_kernel<<<(n + 255) / 256, 256, 0, stream>>>(Wq, wqT, D_MODEL, D_MODEL);
        transpose_to_bf16_kernel<<<(n + 255) / 256, 256, 0, stream>>>(Wk, wkT, D_MODEL, D_MODEL);
        transpose_to_bf16_kernel<<<(n + 255) / 256, 256, 0, stream>>>(Wv, wvT, D_MODEL, D_MODEL);
        transpose_to_bf16_kernel<<<(n + 255) / 256, 256, 0, stream>>>(Wo, woT, D_MODEL, D_MODEL);
        int nf = D_MODEL * FFDIM;
        transpose_to_bf16_kernel<<<(nf + 255) / 256, 256, 0, stream>>>(Wf1, wf1T, D_MODEL, FFDIM);
        transpose_to_bf16_kernel<<<(nf + 255) / 256, 256, 0, stream>>>(Wf2, wf2T, FFDIM, D_MODEL);
    }

    // --- LN1 ---
    layernorm_bf16_kernel<<<MROWS, 256, 0, stream>>>(hidden, g1, b1, normed_bf);

    auto gemm_blocks = [](int M, int Nn) {
        return ((M >> 5) * (Nn >> 6) + 7) / 8;   // 8 waves per 256-thread block
    };

    // --- QKV projections (bf16 out) ---
    gemm_bf16_kernel<false, false, true><<<gemm_blocks(MROWS, D_MODEL), 256, 0, stream>>>(
        normed_bf, wqT, bq, nullptr, q_bf, MROWS, D_MODEL, D_MODEL);
    gemm_bf16_kernel<false, false, true><<<gemm_blocks(MROWS, D_MODEL), 256, 0, stream>>>(
        normed_bf, wkT, bk, nullptr, k_bf, MROWS, D_MODEL, D_MODEL);
    gemm_bf16_kernel<false, false, true><<<gemm_blocks(MROWS, D_MODEL), 256, 0, stream>>>(
        normed_bf, wvT, bv, nullptr, v_bf, MROWS, D_MODEL, D_MODEL);

    // --- V transpose for coalesced WMMA B-fragments in attention ---
    {
        int n = MROWS * D_MODEL;
        v_transpose_kernel<<<(n + 255) / 256, 256, 0, stream>>>(v_bf, vT_bf);
    }

    // --- graph attention ---
    {
        int waves = BATCH * NHEADS * (SEQ_N / 16);   // 4096
        graph_attention_kernel<<<(waves + 7) / 8, 256, 0, stream>>>(
            q_bf, k_bf, vT_bf, adj8, att_bf);
    }

    // --- output projection + residual: hidden2 = hidden + att@Wo + bo ---
    gemm_bf16_kernel<false, true, false><<<gemm_blocks(MROWS, D_MODEL), 256, 0, stream>>>(
        att_bf, woT, bo, hidden, hidden2, MROWS, D_MODEL, D_MODEL);

    // --- LN2 ---
    layernorm_bf16_kernel<<<MROWS, 256, 0, stream>>>(hidden2, g2, b2, normed2_bf);

    // --- FFN ---
    gemm_bf16_kernel<true, false, true><<<gemm_blocks(MROWS, FFDIM), 256, 0, stream>>>(
        normed2_bf, wf1T, bf1, nullptr, ff1_bf, MROWS, D_MODEL, FFDIM);
    gemm_bf16_kernel<false, true, false><<<gemm_blocks(MROWS, D_MODEL), 256, 0, stream>>>(
        ff1_bf, wf2T, bf2, hidden2, (float*)d_out, MROWS, FFDIM, D_MODEL);
}
